// NlEncoder_8074538517122
// MI455X (gfx1250) — matrix operand order repaired and checked by reading, not compiled, not run on hardware
//
#include <hip/hip_runtime.h>
#include <hip/hip_bf16.h>
#include <math.h>

// ---------------------------------------------------------------------------
// CDNA5 / gfx1250 implementation. All large GEMMs run through
// v_wmma_f32_16x16x32_bf16 (f32 -> bf16 conversion fused into LDS staging,
// f32 accumulation). A and B tiles are staged in LDS in fragment-ready
// layouts: every per-lane fragment load is two contiguous 16B ds_load_b128,
// and staging uses bulk register loads + packed b128 LDS stores (no
// per-element predication). Softmax / LN / GRU stay f32 VALU.
// ---------------------------------------------------------------------------

typedef __attribute__((ext_vector_type(16))) __bf16 v16bf;
typedef __attribute__((ext_vector_type(8)))  float  v8f;
typedef __bf16 bf16;

#define E_   256
#define T_   2048
#define NL_  1024
#define FF_  1024
#define NH_  8
#define HD_  32

// ------------------------- block reductions (blockDim==256) ----------------
__device__ __forceinline__ float bsum(float v, float* red) {
  int t = threadIdx.x;
  red[t] = v; __syncthreads();
  for (int s = 128; s > 0; s >>= 1) { if (t < s) red[t] += red[t + s]; __syncthreads(); }
  float r = red[0]; __syncthreads();
  return r;
}
__device__ __forceinline__ float bmax(float v, float* red) {
  int t = threadIdx.x;
  red[t] = v; __syncthreads();
  for (int s = 128; s > 0; s >>= 1) { if (t < s) red[t] = fmaxf(red[t], red[t + s]); __syncthreads(); }
  float r = red[0]; __syncthreads();
  return r;
}

// ------------------------- WMMA GEMM ---------------------------------------
// C[M,N] = act( scale * A[M,K] @ op(B) + bias ),  op(B)=B (KxN) or B^T (NxK)
// Block: 256 threads = 8 waves, computes 64(M) x 128(N) tile, K step 32.
// Wave w: M sub-tile (w&3)*16, N half (w>>2)*64 -> 4 accumulators of 16x16.
#define TM 64
#define TN 128
#define TK 32
#define APITCH (TK + 8)   // 40 bf16 = 80 B rows (16B-aligned row starts)
#define BPITCH (TK + 8)

__global__ __launch_bounds__(256)
void gemm_wmma_kernel(const float* __restrict__ A, const float* __restrict__ B,
                      const float* __restrict__ bias, float* __restrict__ C,
                      int M, int N, int K, int lda, int ldb, int ldc,
                      int transB, float scale, int act)
{
  __shared__ bf16 As [TM][APITCH];   // A tile, k-contiguous per row
  __shared__ bf16 BsT[TN][BPITCH];   // B tile TRANSPOSED: BsT[n][k]

  const int tid  = threadIdx.x;
  const int wave = tid >> 5;
  const int lane = tid & 31;
  const int half = lane >> 4;   // lane group 0..1
  const int l16  = lane & 15;

  const int m0 = blockIdx.y * TM;
  const int n0 = blockIdx.x * TN;
  const int wm = wave & 3;      // M sub-tile 0..3
  const int wn = wave >> 2;     // N half 0..1

  // staging maps (computed once)
  const int a_row = tid >> 2;           // 0..63, 4 threads per row
  const int a_ks  = (tid & 3) * 8;      // 8 contiguous k per thread
  const int b_row = tid & 127;          // 0..127 (n within tile)
  const int b_ks  = (tid >> 7) * 16;    // 16 contiguous k per thread

  const bool a_ok = (m0 + a_row) < M;
  const bool b_ok = (n0 + b_row) < N;

  v8f acc[4] = {v8f{0}, v8f{0}, v8f{0}, v8f{0}};

  for (int k0 = 0; k0 < K; k0 += TK) {
    // ---- stage A tile (64x32): bulk load 8 floats, pack, one b128 store ----
    {
      float va[8];
      if (a_ok) {
        const float* ap = A + (size_t)(m0 + a_row) * lda + (k0 + a_ks);
#pragma unroll
        for (int i = 0; i < 8; ++i) va[i] = ap[i];
      } else {
#pragma unroll
        for (int i = 0; i < 8; ++i) va[i] = 0.f;
      }
      bf16 pk[8];
#pragma unroll
      for (int i = 0; i < 8; ++i) pk[i] = (bf16)va[i];
      *(uint4*)(&As[a_row][a_ks]) = *(const uint4*)pk;
    }
    // ---- stage B tile transposed: bulk load 16 floats, pack, 2 b128 stores --
    {
      float vb[16];
      if (b_ok) {
        if (transB) {
          const float* bp = B + (size_t)(n0 + b_row) * ldb + (k0 + b_ks);
#pragma unroll
          for (int i = 0; i < 16; ++i) vb[i] = bp[i];
        } else {
          const float* bp = B + (size_t)(k0 + b_ks) * ldb + (n0 + b_row);
#pragma unroll
          for (int i = 0; i < 16; ++i) vb[i] = bp[(size_t)i * ldb];
        }
      } else {
#pragma unroll
        for (int i = 0; i < 16; ++i) vb[i] = 0.f;
      }
      bf16 pk[16];
#pragma unroll
      for (int i = 0; i < 16; ++i) pk[i] = (bf16)vb[i];
      *(uint4*)(&BsT[b_row][b_ks])     = *(const uint4*)(pk);
      *(uint4*)(&BsT[b_row][b_ks + 8]) = *(const uint4*)(pk + 8);
    }
    __syncthreads();

    // A fragment: lane-half h holds row (wm*16+l16), K = 8h+(i&7) + 16*(i>>3)
    // -> two contiguous 8-element runs (two ds_load_b128)
    v16bf af;
#pragma unroll
    for (int i = 0; i < 16; ++i) {
      int kk = 8 * half + (i & 7) + ((i >> 3) << 4);
      af[i] = As[wm * 16 + l16][kk];
    }
#pragma unroll
    for (int nt = 0; nt < 4; ++nt) {
      // B fragment: lane n = l16, K = 8*half + (i&7) + 16*(i>>3) -> two 8-element runs
      const int bn = wn * 64 + nt * 16 + l16;
      v16bf bfv;
#pragma unroll
      for (int i = 0; i < 16; ++i)
        bfv[i] = BsT[bn][8 * half + (i & 7) + ((i >> 3) << 4)];
      acc[nt] = __builtin_amdgcn_wmma_f32_16x16x32_bf16(
          false, af, false, bfv, (short)0, acc[nt], false, false);
    }
    __syncthreads();
  }

  // store: C/D layout -> lane n = l16, VGPR r -> M = 8*half + r
#pragma unroll
  for (int nt = 0; nt < 4; ++nt) {
#pragma unroll
    for (int r = 0; r < 8; ++r) {
      int m = m0 + wm * 16 + half * 8 + r;
      int n = n0 + wn * 64 + nt * 16 + l16;
      if (m < M && n < N) {
        float v = acc[nt][r] * scale;
        if (bias) v += bias[n];
        if (act == 1) { // tanh-approx GELU (jax default)
          float u = v;
          v = 0.5f * u * (1.f + tanhf(0.7978845608f * (u + 0.044715f * u * u * u)));
        }
        C[(size_t)m * ldc + n] = v;
      }
    }
  }
}

static inline void gemm(hipStream_t s, const float* A, const float* B,
                        const float* bias, float* C, int M, int N, int K,
                        int lda, int ldb, int ldc, int transB, float scale, int act)
{
  dim3 g((N + TN - 1) / TN, (M + TM - 1) / TM);
  gemm_wmma_kernel<<<g, 256, 0, s>>>(A, B, bias, C, M, N, K, lda, ldb, ldc,
                                     transB, scale, act);
}

// ------------------------- elementwise / softmax kernels --------------------
__global__ __launch_bounds__(256)
void embed_kernel(const int* __restrict__ node, const int* __restrict__ line,
                  const float* __restrict__ tok, const float* __restrict__ tok1,
                  const float* __restrict__ text, const float* __restrict__ mod,
                  const float* __restrict__ chu, float* __restrict__ x)
{
  int t = blockIdx.x, c = threadIdx.x;
  float v;
  if (t < NL_) {
    v = (c < 255) ? tok[(size_t)node[t] * 255 + c] : text[t];
  } else {
    int tt = t - NL_;
    if (c < 254)       v = tok1[(size_t)line[tt] * 254 + c];
    else if (c == 254) v = mod[tt];
    else               v = chu[tt];
  }
  x[(size_t)t * E_ + c] = v;
}

__global__ void e1e2_kernel(const float* __restrict__ Wh, const float* __restrict__ a,
                            float* __restrict__ e1, float* __restrict__ e2)
{
  int t = blockIdx.x * blockDim.x + threadIdx.x;
  if (t >= T_) return;
  float s1 = 0.f, s2 = 0.f;
  for (int c = 0; c < E_; ++c) { float w = Wh[(size_t)t * E_ + c]; s1 += w * a[c]; s2 += w * a[E_ + c]; }
  e1[t] = s1; e2[t] = s2;
}

__global__ __launch_bounds__(256)
void ggann_att_kernel(const float* __restrict__ e1, const float* __restrict__ e2,
                      const float* __restrict__ adj, float* __restrict__ att)
{
  __shared__ float red[256];
  int i = blockIdx.x;
  float e1i = e1[i];
  float mx = -3.0e38f;
  for (int j = threadIdx.x; j < T_; j += 256) {
    float ev = e1i + e2[j];
    ev = ev > 0.f ? ev : 0.2f * ev;                       // leaky_relu alpha=0.2
    float v = (adj[(size_t)i * T_ + j] > 0.f) ? ev : -9.0e15f;
    mx = fmaxf(mx, v);
  }
  mx = bmax(mx, red);
  float sm = 0.f;
  for (int j = threadIdx.x; j < T_; j += 256) {
    float ev = e1i + e2[j];
    ev = ev > 0.f ? ev : 0.2f * ev;
    float v = (adj[(size_t)i * T_ + j] > 0.f) ? ev : -9.0e15f;
    sm += expf(v - mx);
  }
  sm = bsum(sm, red);
  float inv = 1.f / sm;
  for (int j = threadIdx.x; j < T_; j += 256) {
    float ev = e1i + e2[j];
    ev = ev > 0.f ? ev : 0.2f * ev;
    float v = (adj[(size_t)i * T_ + j] > 0.f) ? ev : -9.0e15f;
    att[(size_t)i * T_ + j] = expf(v - mx) * inv;
  }
}

__global__ void gru_kernel(const float* __restrict__ gi, const float* __restrict__ gh,
                           float* __restrict__ h, int elu_out)
{
  int idx = blockIdx.x * blockDim.x + threadIdx.x;
  if (idx >= T_ * E_) return;
  int t = idx >> 8, c = idx & 255;
  const float* gib = gi + (size_t)t * 768;
  const float* ghb = gh + (size_t)t * 768;
  float ir = gib[c], iz = gib[256 + c], inn = gib[512 + c];
  float hr = ghb[c], hz = ghb[256 + c], hn  = ghb[512 + c];
  float r = 1.f / (1.f + expf(-(ir + hr)));
  float z = 1.f / (1.f + expf(-(iz + hz)));
  float n = tanhf(inn + r * hn);
  float hv = h[idx];
  float out = (1.f - z) * n + z * hv;
  if (elu_out) out = out > 0.f ? out : (expf(out) - 1.f);
  h[idx] = out;
}

__global__ __launch_bounds__(256)
void attn_softmax_kernel(float* __restrict__ S, const int* __restrict__ node)
{
  __shared__ float red[256];
  size_t i = blockIdx.x;
  float mx = -3.0e38f;
  for (int j = threadIdx.x; j < T_; j += 256) {
    bool m = (j < NL_) ? (node[j] > 0) : true;
    float v = m ? S[i * T_ + j] : -1.0e9f;
    mx = fmaxf(mx, v);
  }
  mx = bmax(mx, red);
  float sm = 0.f;
  for (int j = threadIdx.x; j < T_; j += 256) {
    bool m = (j < NL_) ? (node[j] > 0) : true;
    float v = m ? S[i * T_ + j] : -1.0e9f;
    sm += expf(v - mx);
  }
  sm = bsum(sm, red);
  float inv = 1.f / sm;
  for (int j = threadIdx.x; j < T_; j += 256) {
    bool m = (j < NL_) ? (node[j] > 0) : true;
    float v = m ? S[i * T_ + j] : -1.0e9f;
    S[i * T_ + j] = expf(v - mx) * inv;
  }
}

__global__ __launch_bounds__(256)
void add_layernorm_kernel(float* __restrict__ x, const float* __restrict__ o,
                          const float* __restrict__ g, const float* __restrict__ b)
{
  __shared__ float red[256];
  size_t i = blockIdx.x; int c = threadIdx.x;
  float v = x[i * E_ + c] + o[i * E_ + c];
  float m = bsum(v, red) * (1.f / E_);
  float d = v - m;
  float var = bsum(d * d, red) * (1.f / E_);
  x[i * E_ + c] = d * rsqrtf(var + 1e-5f) * g[c] + b[c];
}

__global__ void logits_kernel(const float* __restrict__ x, const float* __restrict__ w,
                              const float* __restrict__ b, const int* __restrict__ node,
                              float* __restrict__ logits)
{
  int t = blockIdx.x * blockDim.x + threadIdx.x;
  if (t >= NL_) return;
  float s = 0.f;
  for (int c = 0; c < E_; ++c) s += x[(size_t)t * E_ + c] * w[c];
  s += b[0];
  logits[t] = (node[t] == 2) ? s : -1.0e9f;
}

__global__ __launch_bounds__(256)
void softmax_loss_kernel(const float* __restrict__ logits, const float* __restrict__ res,
                         float* __restrict__ out)
{
  __shared__ float red[256];
  float mx = -3.0e38f;
  for (int j = threadIdx.x; j < NL_; j += 256) mx = fmaxf(mx, logits[j]);
  mx = bmax(mx, red);
  float sm = 0.f;
  for (int j = threadIdx.x; j < NL_; j += 256) sm += expf(logits[j] - mx);
  sm = bsum(sm, red);
  float inv = 1.f / sm;
  float lo = 0.f;
  for (int j = threadIdx.x; j < NL_; j += 256) {
    float p = expf(logits[j] - mx) * inv;
    out[1 + j] = p;                                  // psm
    float pc = fminf(fmaxf(p, 1e-10f), 1.f);
    lo += -logf(pc) * res[j];
  }
  lo = bsum(lo, red);
  if (threadIdx.x == 0) out[0] = lo;                 // loss
}

__global__ void copy_kernel(const float* __restrict__ src, float* __restrict__ dst, int n)
{
  int i = blockIdx.x * blockDim.x + threadIdx.x;
  if (i < n) dst[i] = src[i];
}

// ------------------------- orchestration -----------------------------------
// Input index map (top-level inputs 0..6, then params flattened in dict
// insertion order: tok_emb, tok_emb1, g1{W,a,gru{Wi,Wh,bi,bh}}, g2{...},
// tblocks[0..4]{Wq,bq,Wk,bk,Wv,bv,Wo,bo,ln1_g,ln1_b,W1,b1,W2,b2,ln2_g,ln2_b},
// res2_w, res2_b).
#define FPX(i) ((const float*)d_in[(i)])
#define IPX(i) ((const int*)d_in[(i)])

extern "C" void kernel_launch(void* const* d_in, const int* in_sizes, int n_in,
                              void* d_out, int out_size, void* d_ws, size_t ws_size,
                              hipStream_t stream)
{
  (void)in_sizes; (void)n_in; (void)out_size; (void)ws_size;
  float* ws = (float*)d_ws;

  // workspace layout (floats); total = 9,443,328 floats (~38 MB)
  float* X   = ws;                     // 2048x256 activations (in place)
  float* ATT = ws + 524288;            // 2048x2048 (graph att / per-head scores)
  float* E1  = ws + 4718592;           // 2048
  float* E2  = E1 + 2048;              // 2048
  float* LOG = E2 + 2048;              // 1024 (+pad)
  float* P   = ws + 4724736;           // phase-shared region (4,718,592 floats)
  // GGANN phase aliases
  float* WH = P;                       // 2048x256
  float* HP = P + 524288;              // 2048x256
  float* GI = P + 1048576;             // 2048x768
  float* GH = P + 2621440;             // 2048x768
  // transformer phase aliases
  float* Q  = P;                       // 2048x256
  float* Km = P + 524288;              // 2048x256
  float* V  = P + 1048576;             // 2048x256
  float* AO = P + 1572864;             // 2048x256
  float* T2 = P + 2097152;             // 2048x256
  float* T1 = P + 2621440;             // 2048x1024

  const int*   node = IPX(0);
  const float* adj  = FPX(1);
  const float* res  = FPX(2);
  const float* text = FPX(3);
  const int*   line = IPX(4);
  const float* mod  = FPX(5);
  const float* chu  = FPX(6);

  // 1) embeddings -> X (2048x256)
  embed_kernel<<<T_, 256, 0, stream>>>(node, line, FPX(7), FPX(8), text, mod, chu, X);

  // 2) two GGANN layers (g1 with ELU output, g2 without)
  for (int g = 0; g < 2; ++g) {
    int b = 9 + g * 6;   // W, a, Wi, Wh, bi, bh
    // Wh = h @ W
    gemm(stream, X, FPX(b + 0), nullptr, WH, T_, E_, E_, E_, E_, E_, 0, 1.f, 0);
    e1e2_kernel<<<T_ / 256, 256, 0, stream>>>(WH, FPX(b + 1), E1, E2);
    ggann_att_kernel<<<T_, 256, 0, stream>>>(E1, E2, adj, ATT);
    // h_prime = att @ Wh
    gemm(stream, ATT, WH, nullptr, HP, T_, E_, T_, T_, E_, E_, 0, 1.f, 0);
    // gi = h_prime @ Wi^T + bi ; gh = h @ Wh_gru^T + bh
    gemm(stream, HP, FPX(b + 2), FPX(b + 4), GI, T_, 3 * E_, E_, E_, E_, 3 * E_, 1, 1.f, 0);
    gemm(stream, X,  FPX(b + 3), FPX(b + 5), GH, T_, 3 * E_, E_, E_, E_, 3 * E_, 1, 1.f, 0);
    gru_kernel<<<(T_ * E_) / 256, 256, 0, stream>>>(GI, GH, X, g == 0 ? 1 : 0);
  }

  // 3) 5 transformer blocks
  const float scl = 1.f / sqrtf((float)HD_);
  for (int l = 0; l < 5; ++l) {
    int b = 21 + 16 * l; // Wq,bq,Wk,bk,Wv,bv,Wo,bo,ln1_g,ln1_b,W1,b1,W2,b2,ln2_g,ln2_b
    gemm(stream, X, FPX(b + 0), FPX(b + 1), Q,  T_, E_, E_, E_, E_, E_, 0, 1.f, 0);
    gemm(stream, X, FPX(b + 2), FPX(b + 3), Km, T_, E_, E_, E_, E_, E_, 0, 1.f, 0);
    gemm(stream, X, FPX(b + 4), FPX(b + 5), V,  T_, E_, E_, E_, E_, E_, 0, 1.f, 0);
    for (int h = 0; h < NH_; ++h) {
      // S = Q_h @ K_h^T / sqrt(d)
      gemm(stream, Q + h * HD_, Km + h * HD_, nullptr, ATT,
           T_, T_, HD_, E_, E_, T_, 1, scl, 0);
      attn_softmax_kernel<<<T_, 256, 0, stream>>>(ATT, node);
      // O_h = S @ V_h
      gemm(stream, ATT, V + h * HD_, nullptr, AO + h * HD_,
           T_, HD_, T_, T_, E_, E_, 0, 1.f, 0);
    }
    gemm(stream, AO, FPX(b + 6), FPX(b + 7), T2, T_, E_, E_, E_, E_, E_, 0, 1.f, 0);
    add_layernorm_kernel<<<T_, 256, 0, stream>>>(X, T2, FPX(b + 8), FPX(b + 9));
    gemm(stream, X,  FPX(b + 10), FPX(b + 11), T1, T_, FF_, E_, E_, FF_, FF_, 0, 1.f, 1);
    gemm(stream, T1, FPX(b + 12), FPX(b + 13), T2, T_, E_, FF_, FF_, E_, E_, 0, 1.f, 0);
    add_layernorm_kernel<<<T_, 256, 0, stream>>>(X, T2, FPX(b + 14), FPX(b + 15));
  }

  // 4) head: logits -> softmax -> loss; outputs (loss, psm, x[:1024])
  float* out = (float*)d_out;
  logits_kernel<<<NL_ / 256, 256, 0, stream>>>(X, FPX(101), FPX(102), node, LOG);
  softmax_loss_kernel<<<1, 256, 0, stream>>>(LOG, res, out);
  copy_kernel<<<(NL_ * E_) / 256, 256, 0, stream>>>(X, out + 1 + NL_, NL_ * E_);
}
